// GraphTransformer_91018946937017
// MI455X (gfx1250) — compile-verified, matrix operand order checked by reading, not run on hardware
//
#include <hip/hip_runtime.h>
#include <hip/hip_bf16.h>

// ---------------- problem constants (match reference) ----------------
#define NN   50000
#define EE   800000
#define FIN  128
#define EDIM 16
#define EMB  64
#define HH   2
#define NL   3
#define GG   64
#define HC   (HH * EMB)      // 128
#define EPS  1e-5f
#define INV_SQRT_D 0.125f    // 1/sqrt(64)

typedef __attribute__((ext_vector_type(16))) __bf16 v16bf;
typedef __attribute__((ext_vector_type(8)))  float  v8f;

union FragBF {
    v16bf v;
    unsigned short u[16];
};

__device__ __forceinline__ unsigned short f2bf(float f) {
    unsigned int x = __float_as_uint(f);
    unsigned int r = x + 0x7FFFu + ((x >> 16) & 1u);   // round to nearest even
    return (unsigned short)(r >> 16);
}

// monotone float <-> uint key (for atomicMax-based segment max)
__device__ __forceinline__ unsigned keyenc(float f) {
    unsigned i = __float_as_uint(f);
    return (i & 0x80000000u) ? ~i : (i | 0x80000000u);
}
__device__ __forceinline__ float keydec(unsigned k) {
    unsigned b = (k & 0x80000000u) ? (k ^ 0x80000000u) : ~k;
    return __uint_as_float(b);
}
#define NEGINF_KEY 0x007FFFFFu   // keyenc(-inf)

// ---------------- utility fills ----------------
__global__ void fill_u32(unsigned* p, unsigned val, long long n) {
    long long i = (long long)blockIdx.x * blockDim.x + threadIdx.x;
    long long stride = (long long)gridDim.x * blockDim.x;
    for (; i < n; i += stride) p[i] = val;
}

// ---------------- WMMA GEMM: Y[M,Nout] = act(A[M,K] @ W[Nout,K]^T + bias) ----------------
// One wave computes a 16(M) x 32(N) tile: one A fragment reused across two
// B fragments / two f32 accumulators.  K is compile-time so the loop fully
// unrolls (K/32 * 2 back-to-back v_wmma ops per wave).
// act: 0 = identity, 1 = leaky_relu(0.01)
template <int K>
__global__ void gemm_bf16_wmma(const float* __restrict__ A, const float* __restrict__ W,
                               const float* __restrict__ bias, float* __restrict__ Y,
                               int M, int Nout, int act) {
    int warp = threadIdx.x >> 5;
    int lane = threadIdx.x & 31;
    int mtiles  = (M + 15) >> 4;
    int ntiles2 = Nout >> 5;                 // 32-wide N tiles
    long long wid = (long long)blockIdx.x * (blockDim.x >> 5) + warp;
    if (wid >= (long long)mtiles * ntiles2) return;
    int mt  = (int)(wid / ntiles2);
    int nt2 = (int)(wid % ntiles2);
    int n0  = nt2 * 32;

    int lrow  = lane & 15;
    int khalf = (lane >> 4) << 3;            // 0 for lanes 0-15, 8 for lanes 16-31

    const float* Arow  = A + (size_t)(mt * 16 + lrow) * K;
    const float* Wrow0 = W + (size_t)(n0 + lrow) * K;
    const float* Wrow1 = W + (size_t)(n0 + 16 + lrow) * K;

    v8f acc0 = {};
    v8f acc1 = {};
#pragma unroll
    for (int kt = 0; kt < K; kt += 32) {
        FragBF a, b0, b1;
#pragma unroll
        for (int e = 0; e < 16; e += 2) {
            int vg = e >> 1;
            int kl = ((vg & 4) << 2) + khalf + ((vg & 3) << 1);  // ISA 16-bit A/B layout
            int kk = kt + kl;
            float2 av  = *(const float2*)(Arow + kk);
            float2 bv0 = *(const float2*)(Wrow0 + kk);
            float2 bv1 = *(const float2*)(Wrow1 + kk);
            a.u[e]      = f2bf(av.x);  a.u[e + 1]  = f2bf(av.y);
            b0.u[e]     = f2bf(bv0.x); b0.u[e + 1] = f2bf(bv0.y);
            b1.u[e]     = f2bf(bv1.x); b1.u[e + 1] = f2bf(bv1.y);
        }
        acc0 = __builtin_amdgcn_wmma_f32_16x16x32_bf16(false, a.v, false, b0.v,
                                                       (short)0, acc0, false, false);
        acc1 = __builtin_amdgcn_wmma_f32_16x16x32_bf16(false, a.v, false, b1.v,
                                                       (short)0, acc1, false, false);
    }

    int na = n0 + lrow;
    int nb = n0 + 16 + lrow;
    float bia = bias ? bias[na] : 0.0f;
    float bib = bias ? bias[nb] : 0.0f;
#pragma unroll
    for (int j = 0; j < 8; ++j) {
        int m = mt * 16 + j + khalf;
        float ya = acc0[j] + bia;
        float yb = acc1[j] + bib;
        if (act) {
            ya = (ya > 0.0f) ? ya : 0.01f * ya;
            yb = (yb > 0.0f) ? yb : 0.01f * yb;
        }
        Y[(size_t)m * Nout + na] = ya;
        Y[(size_t)m * Nout + nb] = yb;
    }
}

// ---------------- edge pass A: raw alpha + segment max ----------------
// one wave per edge; lane owns channels [4*lane, 4*lane+4)
__global__ void edge_pass_a(const int* __restrict__ src, const int* __restrict__ dst,
                            const float* __restrict__ ea, const float* __restrict__ ew,
                            const float* __restrict__ qb, const float* __restrict__ kb,
                            float* __restrict__ araw, unsigned* __restrict__ amaxk) {
    __shared__ float sew[HC * EDIM];
    for (int i = threadIdx.x; i < HC * EDIM; i += blockDim.x) sew[i] = ew[i];
    __syncthreads();

    int warp = threadIdx.x >> 5, lane = threadIdx.x & 31;
    long long e = (long long)blockIdx.x * (blockDim.x >> 5) + warp;
    if (e >= EE) return;
    int s = src[e], d = dst[e];

    float eav = (lane < EDIM) ? ea[e * EDIM + lane] : 0.0f;
    int hc0 = lane * 4;
    float emb[4] = {0.f, 0.f, 0.f, 0.f};
#pragma unroll
    for (int i = 0; i < EDIM; ++i) {
        float ai = __shfl(eav, i, 32);
#pragma unroll
        for (int j = 0; j < 4; ++j) emb[j] += ai * sew[(hc0 + j) * EDIM + i];
    }

    float part = 0.f;
#pragma unroll
    for (int j = 0; j < 4; ++j) {
        int hc = hc0 + j;
        part += qb[(size_t)d * HC + hc] * (kb[(size_t)s * HC + hc] + emb[j]);
    }
    // reduce within each 16-lane half (one head per half)
    part += __shfl_xor(part, 1, 32);
    part += __shfl_xor(part, 2, 32);
    part += __shfl_xor(part, 4, 32);
    part += __shfl_xor(part, 8, 32);
    float alpha = part * INV_SQRT_D;

    if ((lane & 15) == 0) {
        int h = lane >> 4;
        araw[e * HH + h] = alpha;
        atomicMax(&amaxk[(size_t)d * HH + h], keyenc(alpha));
    }
}

// ---------------- edge pass B: exp + denominator + unnormalized scatter ----------------
__global__ void edge_pass_b(const int* __restrict__ src, const int* __restrict__ dst,
                            const float* __restrict__ ea, const float* __restrict__ ew,
                            const float* __restrict__ vb, const float* __restrict__ araw,
                            const unsigned* __restrict__ amaxk,
                            float* __restrict__ den, float* __restrict__ attn) {
    __shared__ float sew[HC * EDIM];
    for (int i = threadIdx.x; i < HC * EDIM; i += blockDim.x) sew[i] = ew[i];
    __syncthreads();

    int warp = threadIdx.x >> 5, lane = threadIdx.x & 31;
    long long e = (long long)blockIdx.x * (blockDim.x >> 5) + warp;
    if (e >= EE) return;
    int s = src[e], d = dst[e];

    float eav = (lane < EDIM) ? ea[e * EDIM + lane] : 0.0f;
    int hc0 = lane * 4;
    float emb[4] = {0.f, 0.f, 0.f, 0.f};
#pragma unroll
    for (int i = 0; i < EDIM; ++i) {
        float ai = __shfl(eav, i, 32);
#pragma unroll
        for (int j = 0; j < 4; ++j) emb[j] += ai * sew[(hc0 + j) * EDIM + i];
    }

    int h = lane >> 4;
    float am = keydec(amaxk[(size_t)d * HH + h]);
    float al = __expf(araw[e * HH + h] - am);
    if ((lane & 15) == 0) atomicAdd(&den[(size_t)d * HH + h], al);
#pragma unroll
    for (int j = 0; j < 4; ++j) {
        int hc = hc0 + j;
        float val = (vb[(size_t)s * HC + hc] + emb[j]) * al;
        atomicAdd(&attn[(size_t)d * HC + hc], val);
    }
}

// ---------------- node finalize: normalize + beta gate ----------------
__global__ void node_finalize(float* __restrict__ attn, const float* __restrict__ xr,
                              const float* __restrict__ den, const float* __restrict__ bw) {
    int warp = threadIdx.x >> 5, lane = threadIdx.x & 31;
    long long n = (long long)blockIdx.x * (blockDim.x >> 5) + warp;
    if (n >= NN) return;
    int hc0 = lane * 4;
    float o[4], r[4];
    float part = 0.f;
#pragma unroll
    for (int j = 0; j < 4; ++j) {
        int hc = hc0 + j;
        float dh = den[(size_t)n * HH + (hc >> 6)];
        float ov = attn[(size_t)n * HC + hc];
        ov = (dh > 0.0f) ? ov / dh : 0.0f;
        float rv = xr[(size_t)n * HC + hc];
        o[j] = ov; r[j] = rv;
        part += ov * bw[hc] + rv * bw[HC + hc] + (ov - rv) * bw[2 * HC + hc];
    }
    part += __shfl_xor(part, 1, 32);
    part += __shfl_xor(part, 2, 32);
    part += __shfl_xor(part, 4, 32);
    part += __shfl_xor(part, 8, 32);
    part += __shfl_xor(part, 16, 32);
    float beta = 1.0f / (1.0f + __expf(-part));
#pragma unroll
    for (int j = 0; j < 4; ++j) {
        int hc = hc0 + j;
        attn[(size_t)n * HC + hc] = beta * r[j] + (1.0f - beta) * o[j];
    }
}

// ---------------- batch norm ----------------
__global__ void bn_stats(const float* __restrict__ h, float* __restrict__ sum,
                         float* __restrict__ sq) {
    __shared__ float ssum[256], ssq[256];
    int tid = threadIdx.x;
    int ch = tid & 63;
    int rlane = tid >> 6;     // 0..3
    float s = 0.f, s2 = 0.f;
    for (long long n = (long long)blockIdx.x * 4 + rlane; n < NN; n += (long long)gridDim.x * 4) {
        float v = h[(size_t)n * EMB + ch];
        s += v; s2 += v * v;
    }
    ssum[tid] = s; ssq[tid] = s2;
    __syncthreads();
    if (tid < 64) {
        s  = ssum[tid] + ssum[tid + 64] + ssum[tid + 128] + ssum[tid + 192];
        s2 = ssq[tid]  + ssq[tid + 64]  + ssq[tid + 128]  + ssq[tid + 192];
        atomicAdd(&sum[tid], s);
        atomicAdd(&sq[tid], s2);
    }
}

__global__ void bn_apply(float* __restrict__ h, const float* __restrict__ sum,
                         const float* __restrict__ sq, const float* __restrict__ w,
                         const float* __restrict__ b) {
    long long i = (long long)blockIdx.x * blockDim.x + threadIdx.x;
    long long stride = (long long)gridDim.x * blockDim.x;
    const float invn = 1.0f / (float)NN;
    for (; i < (long long)NN * EMB; i += stride) {
        int c = (int)(i & (EMB - 1));
        float mean = sum[c] * invn;
        float var  = sq[c] * invn - mean * mean;
        float v = h[i];
        h[i] = (v - mean) * rsqrtf(var + EPS) * w[c] + b[c];
    }
}

// ---------------- global pooling ----------------
__global__ void counts_kernel(const int* __restrict__ batch, float* __restrict__ gcnt) {
    long long i = (long long)blockIdx.x * blockDim.x + threadIdx.x;
    long long stride = (long long)gridDim.x * blockDim.x;
    for (; i < NN; i += stride) atomicAdd(&gcnt[batch[i]], 1.0f);
}

__global__ void pool_scatter(const float* __restrict__ h, const int* __restrict__ batch,
                             float* __restrict__ gsum, unsigned* __restrict__ gmaxk) {
    long long i = (long long)blockIdx.x * blockDim.x + threadIdx.x;
    long long stride = (long long)gridDim.x * blockDim.x;
    for (; i < (long long)NN * EMB; i += stride) {
        int n = (int)(i >> 6);
        int c = (int)(i & (EMB - 1));
        int g = batch[n];
        float v = h[i];
        atomicAdd(&gsum[(size_t)g * EMB + c], v);
        atomicMax(&gmaxk[(size_t)g * EMB + c], keyenc(v));
    }
}

__global__ void pool_accum(float* __restrict__ out, const float* __restrict__ gsum,
                           const unsigned* __restrict__ gmaxk, const float* __restrict__ gcnt) {
    int i = blockIdx.x * blockDim.x + threadIdx.x;
    if (i >= GG * 2 * EMB) return;
    int g = i / (2 * EMB);
    int c = i % (2 * EMB);
    float v;
    if (c < EMB) v = keydec(gmaxk[(size_t)g * EMB + c]);
    else         v = gsum[(size_t)g * EMB + (c - EMB)] / fmaxf(gcnt[g], 1.0f);
    out[i] += v;
}

// ---------------- host-side helpers ----------------
static void run_gemm(const float* A, const float* W, const float* bias, float* Y,
                     int M, int K, int Nout, int act, hipStream_t st) {
    long long mtiles  = (M + 15) / 16;
    long long ntiles2 = Nout / 32;
    long long waves   = mtiles * ntiles2;
    int blocks = (int)((waves + 7) / 8);
    if (K == FIN) {
        gemm_bf16_wmma<FIN><<<blocks, 256, 0, st>>>(A, W, bias, Y, M, Nout, act);
    } else {
        gemm_bf16_wmma<EMB><<<blocks, 256, 0, st>>>(A, W, bias, Y, M, Nout, act);
    }
}

static void run_fill(void* p, unsigned val, long long n, hipStream_t st) {
    int blocks = (int)((n + 255) / 256);
    if (blocks > 4096) blocks = 4096;
    fill_u32<<<blocks, 256, 0, st>>>((unsigned*)p, val, n);
}

extern "C" void kernel_launch(void* const* d_in, const int* in_sizes, int n_in,
                              void* d_out, int out_size, void* d_ws, size_t ws_size,
                              hipStream_t stream) {
    const float* x         = (const float*)d_in[0];
    const float* edge_attr = (const float*)d_in[1];
    const int*   eidx      = (const int*)d_in[2];
    const int*   batch     = (const int*)d_in[3];
    const int* src = eidx;
    const int* dst = eidx + EE;

    const float* c1_qw  = (const float*)d_in[4];
    const float* c1_qb  = (const float*)d_in[5];
    const float* c1_kw  = (const float*)d_in[6];
    const float* c1_kb  = (const float*)d_in[7];
    const float* c1_vw  = (const float*)d_in[8];
    const float* c1_vb  = (const float*)d_in[9];
    const float* c1_ew  = (const float*)d_in[10];
    const float* c1_skw = (const float*)d_in[11];
    const float* c1_skb = (const float*)d_in[12];
    const float* c1_bw  = (const float*)d_in[13];
    const float* c1_tw  = (const float*)d_in[14];
    const float* c1_tb  = (const float*)d_in[15];
    const float* c1_bnw = (const float*)d_in[16];
    const float* c1_bnb = (const float*)d_in[17];
    const float* L_qw   = (const float*)d_in[18];
    const float* L_qb   = (const float*)d_in[19];
    const float* L_kw   = (const float*)d_in[20];
    const float* L_kb   = (const float*)d_in[21];
    const float* L_vw   = (const float*)d_in[22];
    const float* L_vb   = (const float*)d_in[23];
    const float* L_ew   = (const float*)d_in[24];
    const float* L_skw  = (const float*)d_in[25];
    const float* L_skb  = (const float*)d_in[26];
    const float* L_bw   = (const float*)d_in[27];
    const float* L_tw   = (const float*)d_in[28];
    const float* L_tb   = (const float*)d_in[29];
    const float* L_bnw  = (const float*)d_in[30];
    const float* L_bnb  = (const float*)d_in[31];

    // ---- workspace layout (floats) ----
    float* base = (float*)d_ws;
    size_t off = 0;
    float* qbuf = base + off; off += (size_t)NN * HC;
    float* kbuf = base + off; off += (size_t)NN * HC;
    float* vbuf = base + off; off += (size_t)NN * HC;
    float* xrbf = base + off; off += (size_t)NN * HC;
    float* attn = base + off; off += (size_t)NN * HC;
    float* h0   = base + off; off += (size_t)NN * EMB;
    float* h1   = base + off; off += (size_t)NN * EMB;
    float* araw = base + off; off += (size_t)EE * HH;
    float* den  = base + off; off += (size_t)NN * HH;
    unsigned* amaxk = (unsigned*)(base + off); off += (size_t)NN * HH;
    float* bnsum = base + off; off += EMB;
    float* bnsq  = base + off; off += EMB;
    float* gsum  = base + off; off += (size_t)GG * EMB;
    float* gcnt  = base + off; off += GG;
    unsigned* gmaxk = (unsigned*)(base + off); off += (size_t)GG * EMB;

    float* out = (float*)d_out;

    // init output + group counts
    run_fill(out, 0u, GG * 2 * EMB, stream);
    run_fill(gcnt, 0u, GG, stream);
    counts_kernel<<<256, 256, 0, stream>>>(batch, gcnt);

    const float* cur = x;
    int Kin = FIN;
    const int edge_blocks = (EE + 7) / 8;
    const int node_blocks = (NN + 7) / 8;

    for (int layer = 0; layer < 1 + NL; ++layer) {
        const float *qw, *qb, *kw, *kb, *vw, *vb, *ew, *skw, *skb, *bwv, *tw, *tb, *bnw, *bnb;
        if (layer == 0) {
            qw = c1_qw; qb = c1_qb; kw = c1_kw; kb = c1_kb; vw = c1_vw; vb = c1_vb;
            ew = c1_ew; skw = c1_skw; skb = c1_skb; bwv = c1_bw;
            tw = c1_tw; tb = c1_tb; bnw = c1_bnw; bnb = c1_bnb;
        } else {
            int i = layer - 1;
            qw = L_qw + (size_t)i * HC * EMB;  qb = L_qb + (size_t)i * HC;
            kw = L_kw + (size_t)i * HC * EMB;  kb = L_kb + (size_t)i * HC;
            vw = L_vw + (size_t)i * HC * EMB;  vb = L_vb + (size_t)i * HC;
            ew = L_ew + (size_t)i * HC * EDIM;
            skw = L_skw + (size_t)i * HC * EMB; skb = L_skb + (size_t)i * HC;
            bwv = L_bw + (size_t)i * 3 * HC;
            tw = L_tw + (size_t)i * EMB * HC;  tb = L_tb + (size_t)i * EMB;
            bnw = L_bnw + (size_t)i * EMB;     bnb = L_bnb + (size_t)i * EMB;
        }

        // node projections via WMMA GEMMs (re-reads hit the 192MB L2)
        run_gemm(cur, qw, qb, qbuf, NN, Kin, HC, 0, stream);
        run_gemm(cur, kw, kb, kbuf, NN, Kin, HC, 0, stream);
        run_gemm(cur, vw, vb, vbuf, NN, Kin, HC, 0, stream);
        run_gemm(cur, skw, skb, xrbf, NN, Kin, HC, 0, stream);

        // attention scratch init
        run_fill(amaxk, 0u, (long long)NN * HH, stream);
        run_fill(den, 0u, (long long)NN * HH, stream);
        run_fill(attn, 0u, (long long)NN * HC, stream);

        edge_pass_a<<<edge_blocks, 256, 0, stream>>>(src, dst, edge_attr, ew,
                                                     qbuf, kbuf, araw, amaxk);
        edge_pass_b<<<edge_blocks, 256, 0, stream>>>(src, dst, edge_attr, ew,
                                                     vbuf, araw, amaxk, den, attn);
        node_finalize<<<node_blocks, 256, 0, stream>>>(attn, xrbf, den, bwv);

        // transf linear + leaky_relu
        float* hnext = (layer & 1) ? h1 : h0;
        run_gemm(attn, tw, tb, hnext, NN, HC, EMB, 1, stream);

        // batch norm
        run_fill(bnsum, 0u, EMB, stream);
        run_fill(bnsq, 0u, EMB, stream);
        bn_stats<<<256, 256, 0, stream>>>(hnext, bnsum, bnsq);
        bn_apply<<<1024, 256, 0, stream>>>(hnext, bnsum, bnsq, bnw, bnb);

        // pooling only for the NL inner layers
        if (layer > 0) {
            run_fill(gsum, 0u, (long long)GG * EMB, stream);
            run_fill(gmaxk, NEGINF_KEY, (long long)GG * EMB, stream);
            pool_scatter<<<2048, 256, 0, stream>>>(hnext, batch, gsum, gmaxk);
            pool_accum<<<(GG * 2 * EMB + 255) / 256, 256, 0, stream>>>(out, gsum, gmaxk, gcnt);
        }

        cur = hnext;
        Kin = EMB;
    }
}
